// NeuralNet_76974403879146
// MI455X (gfx1250) — matrix operand order repaired and checked by reading, not compiled, not run on hardware
//
#include <hip/hip_runtime.h>

typedef __attribute__((ext_vector_type(16))) _Float16 v16h;
typedef __attribute__((ext_vector_type(8)))  float    v8f;

#define B_ROWS   16384
#define IN_DIM   784
#define HID      500
#define HIDP     512
#define OUT_DIM  10
#define STEPS    25
#define BETA     0.95f
#define THRESH   1.0f

#define TILE_M   64
#define THREADS  128          // 4 waves (wave32); wave w owns rows 16w..16w+15
#define KCHUNK   32
#define NKC1     25           // ceil(784/32); K padded to 800 with zeros in staged x
#define XROW     800          // halfs per staged x row (k-contiguous)

// Dynamic LDS layout (bytes)
#define OFF_MEM1   0                         // 64*512 f32 = 131072
#define OFF_CUR1H  131072                    // 64*512 f16 =  65536
#define OFF_W2T    (131072 + 65536)          // 16*512 f16 =  16384
#define OFF_XS     (131072 + 65536 + 16384)  // union: x panel 64*800 f16 = 102400
                                             //        spk1 tile 64*512 f16 = 65536
#define LDS_BYTES  (131072 + 65536 + 16384 + 102400)   // 308 KB <= 320 KB/WGP

static __device__ __forceinline__ v16h load16_cvt(const float* p) {
    const float4* q = (const float4*)p;
    float4 a = q[0], b = q[1], c = q[2], d = q[3];
    v16h r;
    r[0]=(_Float16)a.x;  r[1]=(_Float16)a.y;  r[2]=(_Float16)a.z;  r[3]=(_Float16)a.w;
    r[4]=(_Float16)b.x;  r[5]=(_Float16)b.y;  r[6]=(_Float16)b.z;  r[7]=(_Float16)b.w;
    r[8]=(_Float16)c.x;  r[9]=(_Float16)c.y;  r[10]=(_Float16)c.z; r[11]=(_Float16)c.w;
    r[12]=(_Float16)d.x; r[13]=(_Float16)d.y; r[14]=(_Float16)d.z; r[15]=(_Float16)d.w;
    return r;
}

__global__ __launch_bounds__(THREADS) void snn_fused(
    const float* __restrict__ x,  const float* __restrict__ W1,
    const float* __restrict__ b1, const float* __restrict__ W2,
    const float* __restrict__ b2, float* __restrict__ out)
{
    extern __shared__ char smem[];
    float*    MEM1  = (float*)   (smem + OFF_MEM1);   // [64][512]
    _Float16* CUR1H = (_Float16*)(smem + OFF_CUR1H);  // [64][512]
    _Float16* W2T   = (_Float16*)(smem + OFF_W2T);    // [16][512]
    _Float16* XT    = (_Float16*)(smem + OFF_XS);     // [64][800] row-major (phase 1)
    _Float16* SPK   = (_Float16*)(smem + OFF_XS);     // [64][512] row-major (phase 2)

    const int tid   = threadIdx.x;
    const int lane  = tid & 31;
    const int wv    = tid >> 5;
    const int half  = (lane >> 4) & 1;
    const int ln16  = lane & 15;
    const int mbase = blockIdx.x * TILE_M;

    // ---- phase 0: init mem1, stage W2 (zero-padded 16x512; clamped loads) ----
    for (int i = tid * 4; i < TILE_M * HIDP; i += THREADS * 4) {
        float4 z = {0.f, 0.f, 0.f, 0.f};
        *(float4*)(MEM1 + i) = z;
    }
    for (int i = tid; i < 16 * HIDP; i += THREADS) {
        int n = i >> 9, k = i & 511;
        float v = W2[(n < OUT_DIM ? n : 0) * HID + (k < HID ? k : 0)];
        W2T[i] = (_Float16)((n < OUT_DIM && k < HID) ? v : 0.f);   // pad MUST be 0
    }
    const float b2v = (ln16 < OUT_DIM) ? b2[ln16] : 0.f;

    // ---- stage x panel once, as f16, k-contiguous ----
    // per 32-chunk layout: [k0..31] contiguous; k>=784 -> exact 0
    {
        int row = tid >> 1;
        const float* xr = x + (size_t)(mbase + row) * IN_DIM;
        _Float16* xt = XT + row * XROW;
        for (int kc = (tid & 1); kc < NKC1; kc += 2) {
            int K0 = kc * KCHUNK;
            #pragma unroll
            for (int g = 0; g < 2; ++g) {
                int kb = K0 + g * 16;
                v16h h;
                if (kb + 15 < IN_DIM) {               // group-uniform validity
                    h = load16_cvt(xr + kb);
                } else {
                    #pragma unroll
                    for (int q = 0; q < 16; ++q) h[q] = (_Float16)0.f;
                }
                _Float16* d0 = xt + K0 + (g ? 16 : 0);
                _Float16* d1 = xt + K0 + (g ? 24 : 8);
                #pragma unroll
                for (int c = 0; c < 8; ++c) { d0[c] = h[c]; d1[c] = h[8 + c]; }
            }
        }
    }
    __syncthreads();

    // ---- phase 1: cur1 = x @ W1^T + b1 ----
    // A from k-contiguous LDS; B unconditionally from L2 with address clamping:
    //  - rows n>=500 produce garbage cur1 cols -> killed later by W2T zero pad
    //  - k-group beyond 784 re-reads k=768.. -> killed by a==0 in those slots
    const _Float16* ap = XT + (16 * wv + ln16) * XROW + half * 16;
    for (int ng = 0; ng < HIDP / 64; ++ng) {
        v8f acc0, acc1, acc2, acc3;
        #pragma unroll
        for (int r = 0; r < 8; ++r) { acc0[r]=0.f; acc1[r]=0.f; acc2[r]=0.f; acc3[r]=0.f; }

        const int nbase = ng * 64 + ln16;
        int n0 = nbase;      if (n0 > HID - 1) n0 = HID - 1;
        int n1 = nbase + 16; if (n1 > HID - 1) n1 = HID - 1;
        int n2 = nbase + 32; if (n2 > HID - 1) n2 = HID - 1;
        int n3 = nbase + 48; if (n3 > HID - 1) n3 = HID - 1;
        const float* w0 = W1 + (size_t)n0 * IN_DIM;
        const float* w1 = W1 + (size_t)n1 * IN_DIM;
        const float* w2 = W1 + (size_t)n2 * IN_DIM;
        const float* w3 = W1 + (size_t)n3 * IN_DIM;

        for (int kc = 0; kc < NKC1; ++kc) {
            const int K0 = kc * KCHUNK;
            v16h a;
            #pragma unroll
            for (int i = 0; i < 16; ++i) a[i] = ap[K0 + i];   // 2x ds_load_b128

            int kb = K0 + 16 * half;
            if (kb > IN_DIM - 16) kb = IN_DIM - 16;           // in-bounds; a==0 there

            v16h b0 = load16_cvt(w0 + kb);
            v16h b1f = load16_cvt(w1 + kb);
            v16h b2f = load16_cvt(w2 + kb);
            v16h b3f = load16_cvt(w3 + kb);
            acc0 = __builtin_amdgcn_wmma_f32_16x16x32_f16(false, a, false, b0,  (short)0, acc0, false, false);
            acc1 = __builtin_amdgcn_wmma_f32_16x16x32_f16(false, a, false, b1f, (short)0, acc1, false, false);
            acc2 = __builtin_amdgcn_wmma_f32_16x16x32_f16(false, a, false, b2f, (short)0, acc2, false, false);
            acc3 = __builtin_amdgcn_wmma_f32_16x16x32_f16(false, a, false, b3f, (short)0, acc3, false, false);
        }

        #pragma unroll
        for (int nt = 0; nt < 4; ++nt) {
            const v8f* accp = (nt == 0) ? &acc0 : (nt == 1) ? &acc1 : (nt == 2) ? &acc2 : &acc3;
            int col  = ng * 64 + nt * 16 + ln16;
            int colc = (col < HID) ? col : HID - 1;           // pad cols may be garbage
            float bias = b1[colc];
            #pragma unroll
            for (int r = 0; r < 8; ++r) {
                int rowl = 16 * wv + r + 8 * half;            // C layout: m = r + 8*half
                CUR1H[rowl * HIDP + col] = (_Float16)((*accp)[r] + bias);
            }
        }
    }
    __syncthreads();   // XT region becomes SPK region

    // ---- phase 2: 25 recurrent steps, wave-local (no barriers) ----
    v8f mem2;
    #pragma unroll
    for (int r = 0; r < 8; ++r) mem2[r] = 0.f;

    for (int t = 0; t < STEPS; ++t) {
        // (a) LIF update of mem1 + emit spk1 (f16, row-major)
        {
            float*          mrow = MEM1  + (16 * wv) * HIDP;
            const _Float16* crow = CUR1H + (16 * wv) * HIDP;
            _Float16*       srow = SPK   + (16 * wv) * HIDP;
            for (int i = lane * 4; i < 16 * HIDP; i += 32 * 4) {
                float4 m = *(const float4*)(mrow + i);
                float c0 = (float)crow[i],     c1 = (float)crow[i + 1];
                float c2 = (float)crow[i + 2], c3 = (float)crow[i + 3];
                m.x = BETA * m.x + c0 - (m.x > THRESH ? THRESH : 0.f);
                m.y = BETA * m.y + c1 - (m.y > THRESH ? THRESH : 0.f);
                m.z = BETA * m.z + c2 - (m.z > THRESH ? THRESH : 0.f);
                m.w = BETA * m.w + c3 - (m.w > THRESH ? THRESH : 0.f);
                *(float4*)(mrow + i) = m;
                int w = i & 31;                                // multiple of 4
                int pos = w;
                _Float16* sp = srow + (i & ~31) + pos;
                sp[0] = (_Float16)(m.x > THRESH ? 1.f : 0.f);
                sp[1] = (_Float16)(m.y > THRESH ? 1.f : 0.f);
                sp[2] = (_Float16)(m.z > THRESH ? 1.f : 0.f);
                sp[3] = (_Float16)(m.w > THRESH ? 1.f : 0.f);
            }
        }

        // (b) mem2 = beta*mem2 - reset2 + b2 + spk1 @ W2^T (16 WMMAs, unrolled)
        v8f acc;
        #pragma unroll
        for (int r = 0; r < 8; ++r) {
            float mm = mem2[r];
            acc[r] = BETA * mm - (mm > THRESH ? THRESH : 0.f) + b2v;
        }
        const _Float16* sp2 = SPK + (16 * wv + ln16) * HIDP + half * 16;
        const _Float16* w2p = W2T + ln16 * HIDP + 16 * half;
        #pragma unroll
        for (int kc = 0; kc < HIDP / KCHUNK; ++kc) {
            const int K0 = kc * KCHUNK;
            v16h a, b;
            #pragma unroll
            for (int i = 0; i < 16; ++i) a[i] = sp2[K0 + i];  // 2x ds_load_b128
            #pragma unroll
            for (int i = 0; i < 16; ++i) b[i] = w2p[K0 + i];  // 2x ds_load_b128
            acc = __builtin_amdgcn_wmma_f32_16x16x32_f16(
                false, a, false, b, (short)0, acc, false, false);
        }
        mem2 = acc;

        // (c) record spk2 / mem2
        if (ln16 < OUT_DIM) {
            #pragma unroll
            for (int r = 0; r < 8; ++r) {
                int    gr  = mbase + 16 * wv + r + 8 * half;
                size_t idx = (size_t)t * (B_ROWS * OUT_DIM) + (size_t)gr * OUT_DIM + ln16;
                out[idx] = (acc[r] > THRESH) ? 1.f : 0.f;
                out[(size_t)STEPS * B_ROWS * OUT_DIM + idx] = acc[r];
            }
        }
    }
}

extern "C" void kernel_launch(void* const* d_in, const int* in_sizes, int n_in,
                              void* d_out, int out_size, void* d_ws, size_t ws_size,
                              hipStream_t stream) {
    (void)in_sizes; (void)n_in; (void)d_ws; (void)ws_size; (void)out_size;
    const float* x  = (const float*)d_in[0];
    const float* W1 = (const float*)d_in[1];
    const float* b1 = (const float*)d_in[2];
    const float* W2 = (const float*)d_in[3];
    const float* b2 = (const float*)d_in[4];
    float* out = (float*)d_out;

    hipFuncSetAttribute((const void*)snn_fused,
                        hipFuncAttributeMaxDynamicSharedMemorySize, LDS_BYTES);
    snn_fused<<<dim3(B_ROWS / TILE_M), dim3(THREADS), LDS_BYTES, stream>>>(
        x, W1, b1, W2, b2, out);
}
